// JBF_net_26371099197952
// MI455X (gfx1250) — matrix operand order repaired and checked by reading, not compiled, not run on hardware
//
#include <hip/hip_runtime.h>
#include <stdint.h>

// ---------------------------------------------------------------------------
// JBF net forward for MI455X (gfx1250, wave32, WMMA + async LDS staging).
// Heavy 32->32 3x3x3 convs run as implicit GEMM on v_wmma_f32_16x16x32_f16
// with channels-last f16 activations (K=32 == one WMMA per conv tap).
// B weight table and guide tiles are staged into LDS with
// GLOBAL_LOAD_ASYNC_TO_LDS (ASYNCcnt) when the toolchain exposes it.
// ---------------------------------------------------------------------------

typedef __attribute__((ext_vector_type(16))) _Float16 v16h;
typedef __attribute__((ext_vector_type(8)))  float    v8f;

#define B_   2
#define D_   7
#define H_   256
#define W_   256
#define C_   32
#define NPIX (B_*D_*H_*W_)   // 917504 voxels
#define NIMG (B_*H_*W_)      // 131072 pixels

union ABfrag { v16h v; uint4 q[2]; };

// ---- CDNA5 async global->LDS copies (guarded; fall back to ds stores) -----
#if defined(__has_builtin)
# if __has_builtin(__builtin_amdgcn_global_load_async_to_lds_b128) && \
     __has_builtin(__builtin_amdgcn_global_load_async_to_lds_b32)
#  define HAVE_ASYNC_LDS 1
# endif
#endif
#ifndef HAVE_ASYNC_LDS
# define HAVE_ASYNC_LDS 0
#endif

#if HAVE_ASYNC_LDS
typedef int v4i_gcc __attribute__((vector_size(16)));
typedef __attribute__((address_space(1))) int     g_i32;
typedef __attribute__((address_space(3))) int     l_i32;
typedef __attribute__((address_space(1))) v4i_gcc g_i32v4;
typedef __attribute__((address_space(3))) v4i_gcc l_i32v4;
# define ASYNC_CP_B32(gsrc, ldst)                                            \
    __builtin_amdgcn_global_load_async_to_lds_b32((g_i32*)(gsrc),            \
                                                  (l_i32*)(ldst), 0, 0)
# define ASYNC_CP_B128(gsrc, ldst)                                           \
    __builtin_amdgcn_global_load_async_to_lds_b128((g_i32v4*)(gsrc),         \
                                                   (l_i32v4*)(ldst), 0, 0)
#else
# define ASYNC_CP_B32(gsrc, ldst)  (*(float*)(ldst) = *(const float*)(gsrc))
# define ASYNC_CP_B128(gsrc, ldst) (*(uint4*)(ldst) = *(const uint4*)(gsrc))
#endif

static __device__ __forceinline__ void wait_async_lds()
{
#if HAVE_ASYNC_LDS
# if __has_builtin(__builtin_amdgcn_s_wait_asynccnt)
    __builtin_amdgcn_s_wait_asynccnt(0);
# else
    asm volatile("s_wait_asynccnt 0x0" ::: "memory");
# endif
#endif
}

// ---------------------------------------------------------------------------
// conv1: x (B,1,7,256,256) f32 -> h1 (b,d,h,w,c=32) f16, relu. Direct VALU.
// ---------------------------------------------------------------------------
__global__ void k_conv1(const float* __restrict__ x, const float* __restrict__ w,
                        const float* __restrict__ bias, _Float16* __restrict__ out)
{
    int pix = blockIdx.x * blockDim.x + threadIdx.x;
    if (pix >= NPIX) return;
    int wv = pix % W_; int t = pix / W_;
    int hv = t % H_;   t /= H_;
    int dv = t % D_;   int bv = t / D_;

    float xin[27];
    #pragma unroll
    for (int dd = 0; dd < 3; ++dd)
    #pragma unroll
    for (int dh = 0; dh < 3; ++dh)
    #pragma unroll
    for (int dw = 0; dw < 3; ++dw) {
        int du = dv + dd - 1, hu = hv + dh - 1, wu = wv + dw - 1;
        float v = 0.f;
        if (du >= 0 && du < D_ && hu >= 0 && hu < H_ && wu >= 0 && wu < W_)
            v = x[((bv*D_ + du)*H_ + hu)*W_ + wu];
        xin[(dd*3 + dh)*3 + dw] = v;
    }
    _Float16* dst = out + (size_t)pix * C_;
    for (int co = 0; co < C_; ++co) {
        float acc = bias[co];
        #pragma unroll
        for (int k = 0; k < 27; ++k) acc += w[co*27 + k] * xin[k];
        dst[co] = (_Float16)fmaxf(acc, 0.f);
    }
}

// ---------------------------------------------------------------------------
// weight transform: pwX (co,ci,dd,dh,dw) f32 -> wt[tap][co][ci] f16
// ---------------------------------------------------------------------------
__global__ void k_prepw(const float* __restrict__ w, _Float16* __restrict__ wt)
{
    int i = blockIdx.x * blockDim.x + threadIdx.x;
    if (i >= 27 * C_ * C_) return;
    int ci = i % C_; int t = i / C_;
    int co = t % C_; int tap = t / C_;
    wt[((size_t)tap*C_ + co)*C_ + ci] = (_Float16)w[((size_t)co*C_ + ci)*27 + tap];
}

// ---------------------------------------------------------------------------
// 32->32 3x3x3 SAME conv as implicit GEMM on WMMA f32_16x16x32_f16.
// Block = 256 threads = 8 waves; 64 consecutive W-pixels x 32 out-channels.
// The full 55KB [tap][co][ci] f16 weight table is staged once per block into
// LDS via async global->LDS copies; B fragments are then fed from LDS while
// A fragments stream from global (channels-last, two 16B loads per lane).
// K-loop = 27 taps, each tap exactly one 16x16x32 WMMA.
// ---------------------------------------------------------------------------
__global__ __launch_bounds__(256)
void k_conv_wmma(const _Float16* __restrict__ in, const _Float16* __restrict__ wt,
                 const float* __restrict__ bias, _Float16* __restrict__ out)
{
    __shared__ _Float16 wlds[27 * C_ * C_];          // 55296 B weight table

    // ---- async stage of weight table (3456 x 16B chunks) ----
    for (int i = threadIdx.x; i < 27 * C_ * C_ / 8; i += 256) {
        ASYNC_CP_B128((const uint4*)wt + i, (_Float16*)wlds + (size_t)i * 8);
    }
    wait_async_lds();
    __syncthreads();

    const int lane  = threadIdx.x & 31;
    const int wave  = threadIdx.x >> 5;
    const int mtile = wave & 3;          // 4 M-tiles of 16 pixels
    const int nbase = (wave >> 2) * 16;  // 2 N-halves of 16 channels

    int bid = blockIdx.x;
    const int w0 = (bid & 3) * 64;       // 4 x 64 = 256 = W_
    int t = bid >> 2;
    const int hv = t % H_; t /= H_;
    const int dv = t % D_; const int bv = t / D_;

    const int m    = lane & 15;          // A row (pixel within tile)
    const int ksel = lane >> 4;          // A/B K-half selector
    const int wpix = w0 + mtile*16 + m;

    v8f acc = {0.f,0.f,0.f,0.f,0.f,0.f,0.f,0.f};

    #pragma unroll
    for (int dd = 0; dd < 3; ++dd) {
        const int du = dv + dd - 1;
        #pragma unroll
        for (int dh = 0; dh < 3; ++dh) {
            const int hu = hv + dh - 1;
            #pragma unroll
            for (int dw = 0; dw < 3; ++dw) {
                const int wu  = wpix + dw - 1;
                const int tap = (dd*3 + dh)*3 + dw;

                // A fragment: lanes 0-15 hold K in [8k..8k+8)∪[16+8k..24+8k)
                ABfrag a;
                bool valid = (du >= 0) & (du < D_) & (hu >= 0) & (hu < H_) &
                             (wu >= 0) & (wu < W_);
                if (valid) {
                    const uint4* p = (const uint4*)
                        (in + (size_t)(((bv*D_ + du)*H_ + hu)*W_ + wu) * C_);
                    a.q[0] = p[ksel];       // channels [8*ksel, 8*ksel+8)
                    a.q[1] = p[2 + ksel];   // channels [16+8*ksel, 24+8*ksel)
                } else {
                    a.q[0] = make_uint4(0,0,0,0);
                    a.q[1] = make_uint4(0,0,0,0);
                }

                // B fragment from LDS: lane n holds wt[tap][nbase+n][8k..8k+7]∪[16+8k..24+8k)
                ABfrag b;
                const uint4* q = (const uint4*)
                    ((const _Float16*)wlds + ((size_t)tap*C_ + (nbase + m))*C_
                                           + ksel*8);
                b.q[0] = q[0];
                b.q[1] = q[2];

                acc = __builtin_amdgcn_wmma_f32_16x16x32_f16(
                          false, a.v, false, b.v, (short)0, acc, false, false);
            }
        }
    }

    // Epilogue: lane holds N = lane&15, M rows mbase..mbase+7 in acc[0..7]
    const int n     = lane & 15;
    const int co    = nbase + n;
    const float bco = bias[co];
    const int mbase = (lane >> 4) * 8;
    #pragma unroll
    for (int r = 0; r < 8; ++r) {
        const int wpo = w0 + mtile*16 + mbase + r;
        size_t pix = (size_t)(((bv*D_ + dv)*H_ + hv)*W_ + wpo);
        out[pix*C_ + co] = (_Float16)fmaxf(acc[r] + bco, 0.f);
    }
}

// ---------------------------------------------------------------------------
// conv4 (32->1) + skip: guide = conv(h3) + x ; also IS the "prior" output.
// ---------------------------------------------------------------------------
__global__ void k_guide(const _Float16* __restrict__ h3, const float* __restrict__ w4,
                        const float* __restrict__ b4, const float* __restrict__ x,
                        float* __restrict__ prior)
{
    int pix = blockIdx.x * blockDim.x + threadIdx.x;
    if (pix >= NPIX) return;
    int wv = pix % W_; int t = pix / W_;
    int hv = t % H_;   t /= H_;
    int dv = t % D_;   int bv = t / D_;

    float acc = b4[0];
    #pragma unroll
    for (int dd = 0; dd < 3; ++dd)
    #pragma unroll
    for (int dh = 0; dh < 3; ++dh)
    #pragma unroll
    for (int dw = 0; dw < 3; ++dw) {
        int du = dv + dd - 1, hu = hv + dh - 1, wu = wv + dw - 1;
        if (du >= 0 && du < D_ && hu >= 0 && hu < H_ && wu >= 0 && wu < W_) {
            const _Float16* p = h3 + (size_t)(((bv*D_+du)*H_+hu)*W_+wu) * C_;
            const int tap = (dd*3 + dh)*3 + dw;
            for (int ci = 0; ci < C_; ++ci)
                acc += (float)p[ci] * w4[ci*27 + tap];
        }
    }
    prior[pix] = acc + x[pix];
}

// ---------------------------------------------------------------------------
// domain_k for all 4 blocks: coeff_net(spat_kernel) (tiny, one block).
// ---------------------------------------------------------------------------
__global__ void k_dom(const float* __restrict__ spat,
                      const float* __restrict__ dw1, const float* __restrict__ db1,
                      const float* __restrict__ dw2, const float* __restrict__ db2,
                      float* __restrict__ dom)
{
    int q = threadIdx.x;
    if (q >= 27) return;
    const int qd = q/9, qh = (q/3)%3, qw = q%3;
    for (int blk = 0; blk < 4; ++blk) {
        float acc = db2[blk];
        for (int co = 0; co < 8; ++co) {
            for (int t2 = 0; t2 < 27; ++t2) {
                const int pd = qd + t2/9, ph = qh + (t2/3)%3, pw = qw + t2%3;
                float h = db1[blk*8 + co];
                #pragma unroll
                for (int t = 0; t < 27; ++t) {
                    int ad = pd + t/9, ah = ph + (t/3)%3, aw2 = pw + t%3;
                    h += dw1[(blk*8 + co)*27 + t] * spat[(ad*7 + ah)*7 + aw2];
                }
                acc += dw2[(blk*8 + co)*27 + t2] * fmaxf(h, 0.f);
            }
        }
        dom[blk*27 + q] = fmaxf(acc, 0.f);
    }
}

// ---------------------------------------------------------------------------
// range_k for all 4 blocks. Per block of 64 consecutive W-pixels:
//   1) async-stage the raw guide tile (7 d x 7 h x 70 w, zero-padded) to LDS
//   2) each thread builds its |patch - center| 7^3 in LDS (f16)
//   3) register-resident coeff-net: 8ch 5^3 -> 27 outputs, x4 blocks.
// ---------------------------------------------------------------------------
__global__ __launch_bounds__(64)
void k_range(const float* __restrict__ guide,
             const float* __restrict__ rw1, const float* __restrict__ rb1,
             const float* __restrict__ rw2, const float* __restrict__ rb2,
             float* __restrict__ rk)
{
    __shared__ float    sraw[7][7][72];   // 14112 B raw guide tile
    __shared__ _Float16 sp[64][344];      // 44032 B per-thread |patch-center|

    const int tid = threadIdx.x;
    const int n = blockIdx.x * 64 + tid;  // grid is exact: n < NIMG
    const int wv = n % W_; int t = n / W_;
    const int hv = t % H_; const int bv = t / H_;
    const int w0 = wv - tid;              // 64-aligned tile base

    // ---- stage raw tile: (d, i, j) -> guide[d, hv+i-3, w0+j-3], 0 outside --
    for (int idx = tid; idx < 7*7*70; idx += 64) {
        int j = idx % 70; int r = idx / 70;
        int i = r % 7;    int d = r / 7;
        int hu = hv + i - 3;
        int wu = w0 + j - 3;
        float* dst = &sraw[d][i][j];
        if (hu >= 0 && hu < H_ && wu >= 0 && wu < W_) {
            ASYNC_CP_B32(&guide[((bv*D_ + d)*H_ + hu)*W_ + wu], dst);
        } else {
            *dst = 0.f;
        }
    }
    wait_async_lds();
    __syncthreads();

    // ---- per-thread |patch - center| (own LDS row, no barrier needed) -----
    const float center = sraw[3][3][tid + 3];
    for (int d = 0; d < 7; ++d)
        for (int i = 0; i < 7; ++i)
            for (int j = 0; j < 7; ++j)
                sp[tid][(d*7 + i)*7 + j] =
                    (_Float16)fabsf(sraw[d][i][tid + j] - center);

    for (int blk = 0; blk < 4; ++blk) {
        for (int q = 0; q < 27; ++q) {
            const int qd = q/9, qh = (q/3)%3, qw = q%3;
            float acc = rb2[blk];
            for (int co = 0; co < 8; ++co) {
                for (int t2 = 0; t2 < 27; ++t2) {
                    const int pd = qd + t2/9, ph = qh + (t2/3)%3, pw = qw + t2%3;
                    float h = rb1[blk*8 + co];
                    #pragma unroll
                    for (int tt = 0; tt < 27; ++tt) {
                        int ad = pd + tt/9, ah = ph + (tt/3)%3, aw2 = pw + tt%3;
                        h += rw1[(blk*8 + co)*27 + tt] *
                             (float)sp[tid][(ad*7 + ah)*7 + aw2];
                    }
                    acc += rw2[(blk*8 + co)*27 + t2] * fmaxf(h, 0.f);
                }
            }
            rk[((size_t)blk*NIMG + n)*27 + q] = fmaxf(acc, 0.f);
        }
    }
}

// ---------------------------------------------------------------------------
// xc init: x[:,:,2:5] (no relu on first iteration).
// ---------------------------------------------------------------------------
__global__ void k_init(const float* __restrict__ x, float* __restrict__ xc)
{
    int i = blockIdx.x * blockDim.x + threadIdx.x;
    if (i >= B_*3*H_*W_) return;
    int wv = i % W_; int t = i / W_;
    int hv = t % H_; t /= H_;
    int dd = t % 3;  int bv = t / 3;
    xc[i] = x[((bv*D_ + (2 + dd))*H_ + hv)*W_ + wv];
}

// ---------------------------------------------------------------------------
// bilateral apply (per block): wt = dom*rk, normalize, weighted 3^3 sum of xc.
// ---------------------------------------------------------------------------
__global__ void k_y(const float* __restrict__ rk, const float* __restrict__ dom,
                    const float* __restrict__ xc, const float* __restrict__ x,
                    float* __restrict__ ytmp, float* __restrict__ rbuf, int blk)
{
    int n = blockIdx.x * blockDim.x + threadIdx.x;
    if (n >= NIMG) return;
    const int wv = n % W_; int t = n / W_;
    const int hv = t % H_; const int bv = t / H_;

    const float* rkp = rk + ((size_t)blk*NIMG + n)*27;
    float wt[27]; float s = 0.f;
    #pragma unroll
    for (int q = 0; q < 27; ++q) { wt[q] = dom[blk*27 + q] * rkp[q]; s += wt[q]; }
    const float inv = 1.f / (s + 1e-6f);

    float y = 0.f;
    #pragma unroll
    for (int dd = 0; dd < 3; ++dd)
    #pragma unroll
    for (int hh = 0; hh < 3; ++hh)
    #pragma unroll
    for (int ww = 0; ww < 3; ++ww) {
        int hu = hv + hh - 1, wu = wv + ww - 1;
        float v = 0.f;
        if (hu >= 0 && hu < H_ && wu >= 0 && wu < W_)
            v = xc[((bv*3 + dd)*H_ + hu)*W_ + wu];
        y += wt[(dd*3 + hh)*3 + ww] * v;
    }
    y = fmaxf(y * inv, 0.f);
    ytmp[n] = y;
    rbuf[n] = y - x[((bv*D_ + 3)*H_ + hv)*W_ + wv];
}

__global__ void k_alfa(const float* __restrict__ ytmp, const float* __restrict__ rbuf,
                       const float* __restrict__ aw, const float* __restrict__ ab,
                       const float* __restrict__ x, float* __restrict__ xc,
                       float* __restrict__ yout, int blk)
{
    int n = blockIdx.x * blockDim.x + threadIdx.x;
    if (n >= NIMG) return;
    const int wv = n % W_; int t = n / W_;
    const int hv = t % H_; const int bv = t / H_;

    float alfa = ab[blk];
    #pragma unroll
    for (int ih = 0; ih < 3; ++ih)
    #pragma unroll
    for (int iw = 0; iw < 3; ++iw) {
        int hu = hv + ih - 1, wu = wv + iw - 1;
        float v = 0.f;
        if (hu >= 0 && hu < H_ && wu >= 0 && wu < W_)
            v = rbuf[(bv*H_ + hu)*W_ + wu];
        alfa += aw[blk*9 + ih*3 + iw] * v;
    }
    float y2 = fmaxf(ytmp[n] + alfa * rbuf[n], 0.f);
    yout[n] = y2;
    // xc <- relu(concat(x[2], y2, x[4]))  (x-slice relus are idempotent)
    xc[((bv*3 + 0)*H_ + hv)*W_ + wv] = fmaxf(x[((bv*D_ + 2)*H_ + hv)*W_ + wv], 0.f);
    xc[((bv*3 + 1)*H_ + hv)*W_ + wv] = y2;
    xc[((bv*3 + 2)*H_ + hv)*W_ + wv] = fmaxf(x[((bv*D_ + 4)*H_ + hv)*W_ + wv], 0.f);
}

// ---------------------------------------------------------------------------
extern "C" void kernel_launch(void* const* d_in, const int* in_sizes, int n_in,
                              void* d_out, int out_size, void* d_ws, size_t ws_size,
                              hipStream_t stream)
{
    (void)in_sizes; (void)n_in; (void)out_size; (void)ws_size;
    const float* x    = (const float*)d_in[0];
    const float* spat = (const float*)d_in[1];
    const float* pw1  = (const float*)d_in[2];
    const float* pb1  = (const float*)d_in[3];
    const float* pw2  = (const float*)d_in[4];
    const float* pb2  = (const float*)d_in[5];
    const float* pw3  = (const float*)d_in[6];
    const float* pb3  = (const float*)d_in[7];
    const float* pw4  = (const float*)d_in[8];
    const float* pb4  = (const float*)d_in[9];
    const float* dw1  = (const float*)d_in[10];
    const float* db1  = (const float*)d_in[11];
    const float* dw2  = (const float*)d_in[12];
    const float* db2  = (const float*)d_in[13];
    const float* rw1  = (const float*)d_in[14];
    const float* rb1  = (const float*)d_in[15];
    const float* rw2  = (const float*)d_in[16];
    const float* rb2  = (const float*)d_in[17];
    const float* aw   = (const float*)d_in[18];
    const float* ab   = (const float*)d_in[19];

    float* out       = (float*)d_out;
    float* out_xc    = out;                              // (2,1,3,256,256)
    float* out_prior = out + (size_t)B_*3*H_*W_;         // (2,1,7,256,256) == guide
    float* out_y0    = out_prior + (size_t)NPIX;         // 4 x (2,1,1,256,256)

    char* ws = (char*)d_ws;
    size_t off = 0;
    auto alloc = [&](size_t bytes) -> char* {
        char* p = ws + off; off += (bytes + 255) & ~(size_t)255; return p;
    };
    _Float16* hA  = (_Float16*)alloc((size_t)NPIX * C_ * sizeof(_Float16)); // h1 / h3
    _Float16* hB  = (_Float16*)alloc((size_t)NPIX * C_ * sizeof(_Float16)); // h2
    _Float16* wt2 = (_Float16*)alloc((size_t)27 * C_ * C_ * sizeof(_Float16));
    _Float16* wt3 = (_Float16*)alloc((size_t)27 * C_ * C_ * sizeof(_Float16));
    float* rk   = (float*)alloc((size_t)4 * NIMG * 27 * sizeof(float));
    float* dom  = (float*)alloc((size_t)4 * 27 * sizeof(float));
    float* ytmp = (float*)alloc((size_t)NIMG * sizeof(float));
    float* rbuf = (float*)alloc((size_t)NIMG * sizeof(float));

    // prior network
    k_conv1<<<(NPIX + 255)/256, 256, 0, stream>>>(x, pw1, pb1, hA);
    k_prepw<<<(27*C_*C_ + 255)/256, 256, 0, stream>>>(pw2, wt2);
    k_prepw<<<(27*C_*C_ + 255)/256, 256, 0, stream>>>(pw3, wt3);
    k_conv_wmma<<<NPIX/64, 256, 0, stream>>>(hA, wt2, pb2, hB);   // conv2 (WMMA)
    k_conv_wmma<<<NPIX/64, 256, 0, stream>>>(hB, wt3, pb3, hA);   // conv3 (WMMA)
    k_guide<<<(NPIX + 255)/256, 256, 0, stream>>>(hA, pw4, pb4, x, out_prior);

    // coefficient nets (block-independent, computed once)
    k_dom<<<1, 32, 0, stream>>>(spat, dw1, db1, dw2, db2, dom);
    k_range<<<NIMG/64, 64, 0, stream>>>(out_prior, rw1, rb1, rw2, rb2, rk);

    // sequential bilateral blocks
    k_init<<<(B_*3*H_*W_ + 255)/256, 256, 0, stream>>>(x, out_xc);
    for (int blk = 0; blk < 4; ++blk) {
        k_y<<<(NIMG + 255)/256, 256, 0, stream>>>(rk, dom, out_xc, x, ytmp, rbuf, blk);
        k_alfa<<<(NIMG + 255)/256, 256, 0, stream>>>(ytmp, rbuf, aw, ab, x, out_xc,
                                                     out_y0 + (size_t)blk*NIMG, blk);
    }
}
